// ARCTransformer_68478958567578
// MI455X (gfx1250) — matrix operand order repaired and checked by reading, not compiled, not run on hardware
//
#include <hip/hip_runtime.h>
#include <hip/hip_bf16.h>
#include <math.h>

// ---------------- problem constants ----------------
constexpr int kG  = 20;
constexpr int kT  = 2005;   // tokens
constexpr int kTP = 2016;   // padded to multiple of 16
constexpr int kB  = 2;
constexpr int kD  = 512;
constexpr int kF  = 2048;
constexpr int kV  = 12;
constexpr int kL  = 4;
constexpr int kH  = 8;
constexpr int kDH = 64;

typedef __attribute__((ext_vector_type(16))) _Float16 v16h;
typedef __attribute__((ext_vector_type(8)))  float    v8f;
typedef __attribute__((ext_vector_type(4)))  unsigned v4u;

// ---------------- WMMA helpers (CDNA5 wave32) ----------------
__device__ __forceinline__ v8f wmma_f16(v16h a, v16h b, v8f c) {
  // D = A(16x32 f16) * B(32x16 f16) + C(16x16 f32)
  return __builtin_amdgcn_wmma_f32_16x16x32_f16(
      /*neg_a=*/false, a, /*neg_b=*/false, b,
      /*c_mod=*/(short)0, c, /*reuse_a=*/false, /*reuse_b=*/false);
}

// A fragment: 16x32 f16 tile, row-major, leading dim lda (16B-aligned rows).
// Lane layout: m = lane&15; K-half = lane>>4. Two 16-byte contiguous chunks.
__device__ __forceinline__ v16h load_a_frag(const _Float16* A, int lda, int lane) {
  int m = lane & 15, kh = lane >> 4;
  const _Float16* r = A + (size_t)m * lda + kh * 8;
  union { v16h v; v4u q[2]; } f;
  f.q[0] = *(const v4u*)(r);        // K = kh*8 .. kh*8+7
  f.q[1] = *(const v4u*)(r + 16);   // K = 16+kh*8 .. 16+kh*8+7
  return f.v;
}

// B fragment from a TRANSPOSED operand: B[k][n] = Bt[n*ld + k].
// Lane layout: n = lane&15 selects row of Bt; K-half = lane>>4: K = kh*8..kh*8+7 and 16+kh*8..16+kh*8+7.
// 8 contiguous f16 at r and 8 at r+16 per lane -> two 16-byte loads.
__device__ __forceinline__ v16h load_bT_frag(const _Float16* Bt, int ld, int lane) {
  int n = lane & 15, kh = lane >> 4;
  const _Float16* r = Bt + (size_t)n * ld + kh * 8;
  union { v16h v; v4u q[2]; } f;
  f.q[0] = *(const v4u*)(r);
  f.q[1] = *(const v4u*)(r + 16);
  return f.v;
}

__device__ __forceinline__ float gelu_tanh(float x) {
  float x3 = x * x * x;
  return 0.5f * x * (1.f + tanhf(0.7978845608028654f * (x + 0.044715f * x3)));
}

// ---------------- coordinates (packed per token) ----------------
// bits: row[0:5) col[5:10) sep[10] type[11:13) drow[13:18) dcol[18:24) did+1[24:26)
__global__ void coords_kernel(int* __restrict__ coords) {
  for (int t = blockIdx.x * blockDim.x + threadIdx.x; t < kTP; t += blockDim.x * gridDim.x) {
    int c;
    if (t >= kT) {
      c = (1 << 10);  // sep=1, did+1=0 -> never contributes
    } else {
      int within = t % 401;
      int is_sep = (within == 400) ? 1 : 0;
      int cell = within < 399 ? within : 399;
      int row = cell / kG, col = cell % kG;
      int in_demos = (t < 1604) ? 1 : 0;
      int wd = t % 802;
      int in_x = in_demos && (wd < 400);
      int in_y = in_demos && (wd > 400) && (wd < 801);
      int wt = t - 1604;
      int in_tx = (!in_demos) && (wt < 400);
      int ttype = in_x ? 0 : (in_y ? 1 : (in_tx ? 2 : 3));
      int did = is_sep ? -1 : (in_demos ? (t / 802) : 2);
      int cy = wd - 401;
      int drow = in_x ? (wd / kG) : (in_y ? (cy / kG) : (in_tx ? (wt / kG) : 0));
      int dcol = in_x ? (wd % kG) : (in_y ? ((cy % kG) + kG + 1) : (in_tx ? (wt % kG) : 0));
      c = row | (col << 5) | (is_sep << 10) | (ttype << 11) | (drow << 13) |
          (dcol << 18) | ((did + 1) << 24);
    }
    coords[t] = c;
  }
}

__device__ __forceinline__ float rel_bias(int cq, int ck, int h,
                                          const float* __restrict__ br,
                                          const float* __restrict__ bd) {
  int sq = (cq >> 10) & 1, sk = (ck >> 10) & 1;
  if (sq | sk) return 0.f;
  int rq = cq & 31, colq = (cq >> 5) & 31;
  int rk = ck & 31, colk = (ck >> 5) & 31;
  int dr = min(max(rq - rk, -(kG - 1)), kG - 1) + (kG - 1);
  int dc = min(max(colq - colk, -(kG - 1)), kG - 1) + (kG - 1);
  float v = br[(dr * (2 * kG - 1) + dc) * kH + h];
  int dq = ((cq >> 24) & 3) - 1, dk = ((ck >> 24) & 3) - 1;
  if (dq == dk && dq >= 0) {
    int drq = (cq >> 13) & 31, drk = (ck >> 13) & 31;
    int dcq = (cq >> 18) & 63, dck = (ck >> 18) & 63;
    int ddr = min(max(drq - drk, -(kG - 1)), kG - 1) + (kG - 1);
    int ddc = min(max(dcq - dck, -2 * kG), 2 * kG) + 2 * kG;
    v += bd[(ddr * (4 * kG + 1) + ddc) * kH + h];
  }
  return v;
}

// ---------------- embedding ----------------
__global__ void embed_kernel(const int* __restrict__ tokens,
                             const float* __restrict__ tok_emb,
                             const float* __restrict__ row_emb,
                             const float* __restrict__ col_emb,
                             const float* __restrict__ type_emb,
                             const int* __restrict__ coords,
                             float* __restrict__ x) {
  int bt = blockIdx.x, lane = threadIdx.x;
  int b = bt / kTP, t = bt % kTP;
  float* xr = x + (size_t)bt * kD;
  if (t >= kT) {
#pragma unroll
    for (int i = 0; i < 16; ++i) xr[lane + i * 32] = 0.f;
    return;
  }
  int c = coords[t];
  int row = c & 31, col = (c >> 5) & 31, sep = (c >> 10) & 1, tt = (c >> 11) & 3;
  int tok = tokens[b * kT + t];
  const float* te = tok_emb + (size_t)tok * kD;
  const float* ty = type_emb + (size_t)tt * kD;
  const float* re = row_emb + (size_t)row * kD;
  const float* ce = col_emb + (size_t)col * kD;
  float pm = sep ? 0.f : 1.f;
#pragma unroll
  for (int i = 0; i < 16; ++i) {
    int d = lane + i * 32;
    xr[d] = te[d] + ty[d] + (re[d] + ce[d]) * pm;
  }
}

// ---------------- layernorm (one wave per row, D=512) ----------------
__global__ void ln_kernel(const float* __restrict__ x, const float* __restrict__ s,
                          const float* __restrict__ bb, _Float16* __restrict__ out) {
  int r = blockIdx.x, lane = threadIdx.x;
  const float* xr = x + (size_t)r * kD;
  float v[16];
  float sum = 0.f;
#pragma unroll
  for (int i = 0; i < 16; ++i) { v[i] = xr[lane + i * 32]; sum += v[i]; }
#pragma unroll
  for (int o = 16; o >= 1; o >>= 1) sum += __shfl_xor(sum, o, 32);
  float mean = sum * (1.f / 512.f);
  float vs = 0.f;
#pragma unroll
  for (int i = 0; i < 16; ++i) { float d = v[i] - mean; vs += d * d; }
#pragma unroll
  for (int o = 16; o >= 1; o >>= 1) vs += __shfl_xor(vs, o, 32);
  float inv = rsqrtf(vs * (1.f / 512.f) + 1e-5f);
  _Float16* orow = out + (size_t)r * kD;
#pragma unroll
  for (int i = 0; i < 16; ++i) {
    int d = lane + i * 32;
    orow[d] = (_Float16)((v[i] - mean) * inv * s[d] + bb[d]);
  }
}

// ------------- fp32 -> f16 with per-layer transpose: (L,K,N) -> (L,N,K) -------------
__global__ void cvtT_kernel(const float* __restrict__ src, _Float16* __restrict__ dst,
                            int K, int N, int total) {
  int i = blockIdx.x * blockDim.x + threadIdx.x;
  if (i >= total) return;
  int kn = K * N;
  int l = i / kn, r = i % kn;
  int k = r / N, n = r % N;
  dst[(size_t)l * kn + (size_t)n * K + k] = (_Float16)src[i];
}

// ---- V transpose per head: qkv v-part (b,t,h,d) -> Vt[b][h][d][t] (f16) ----
__global__ void vtrans_kernel(const _Float16* __restrict__ qkv, _Float16* __restrict__ Vt) {
  int bt = blockIdx.x;
  int b = bt / kTP, t = bt % kTP;
  const _Float16* vrow = qkv + (size_t)bt * 3 * kD + 2 * kD;
  for (int i = threadIdx.x; i < kD; i += blockDim.x) {
    int h = i >> 6, d = i & 63;
    Vt[(((size_t)b * kH + h) * kDH + d) * kTP + t] = vrow[i];
  }
}

// ------- generic WMMA GEMM: C = act(A*Bt^T + bias) (+resid), 16x64 per wave -------
// A: MxK f16 row-major; Bt: NxK f16 row-major (pre-transposed weights).
template <int ACT>
__global__ void gemm16(const _Float16* __restrict__ A, int lda,
                       const _Float16* __restrict__ Bt, int K,
                       const float* __restrict__ bias,
                       const float* resid, float* Cf, _Float16* Ch,
                       int ldc) {
  int lane = threadIdx.x;
  int m0 = blockIdx.x * 16, n0 = blockIdx.y * 64;
  const _Float16* Ab = A + (size_t)m0 * lda;
  const _Float16* Bb = Bt + (size_t)n0 * K;
  v8f acc[4] = {};
  for (int k = 0; k < K; k += 32) {
    if (k + 32 < K)
      __builtin_prefetch(Ab + (size_t)(lane & 15) * lda + k + 32, 0, 1);
    v16h a = load_a_frag(Ab + k, lda, lane);
#pragma unroll
    for (int j = 0; j < 4; ++j) {
      v16h b = load_bT_frag(Bb + (size_t)(j * 16) * K + k, K, lane);
      acc[j] = wmma_f16(a, b, acc[j]);
    }
  }
  int n = lane & 15, half = lane >> 4;
#pragma unroll
  for (int j = 0; j < 4; ++j) {
    int nc = n0 + j * 16 + n;
    float bn = bias ? bias[nc] : 0.f;
#pragma unroll
    for (int r = 0; r < 8; ++r) {
      int m = m0 + r + half * 8;
      size_t idx = (size_t)m * ldc + nc;
      float vv = acc[j][r] + bn;
      if (ACT == 1) vv = gelu_tanh(vv);
      if (resid) vv += resid[idx];
      if (Cf) Cf[idx] = vv;
      if (Ch) Ch[idx] = (_Float16)vv;
    }
  }
}

// ---------------- flash attention (one wave per (b, h, 16-query tile)) ----------------
// S^T = K*Q^T per 16-key subtile via WMMA; online softmax per query (lane column);
// P staged transposed into LDS; O^T = Vt * P^T via WMMA (4 dh-tiles of 16).
__global__ void attn_kernel(const _Float16* __restrict__ qkv,
                            const _Float16* __restrict__ Vt,
                            const int* __restrict__ coords,
                            const float* __restrict__ bias_rel,
                            const float* __restrict__ bias_demo,
                            _Float16* __restrict__ out) {
  __shared__ __align__(16) _Float16 lds_p[16 * 32];  // [query][key]
  int lane = threadIdx.x;
  int q0 = blockIdx.x * 16;
  int h = blockIdx.y;
  int b = blockIdx.z;
  int n = lane & 15, half = lane >> 4;
  size_t rowbase = (size_t)b * kTP * 3 * kD;
  const int ldq = 3 * kD;
  const _Float16* Q = qkv + rowbase + (size_t)q0 * ldq + h * kDH;  // Q[n][d]
  const _Float16* Vth = Vt + ((size_t)b * kH + h) * kDH * kTP;     // Vt[d][t]
  int cq = coords[q0 + n];
  const float scale = 0.125f;  // 1/sqrt(64)

  v8f acc[4] = {};
  float m_run = -INFINITY, l_run = 0.f;

  for (int k0 = 0; k0 < kTP; k0 += 32) {
    float p[16];
    float mloc = -INFINITY;
#pragma unroll
    for (int s = 0; s < 2; ++s) {
      int kb = k0 + s * 16;
      const _Float16* Kp = qkv + rowbase + (size_t)kb * ldq + kD + h * kDH;
      v8f sc = {};
#pragma unroll
      for (int dstep = 0; dstep < 2; ++dstep) {
        v16h a = load_a_frag(Kp + dstep * 32, ldq, lane);   // K rows (keys x d)
        v16h bq = load_bT_frag(Q + dstep * 32, ldq, lane);  // Q^T (d x queries)
        sc = wmma_f16(a, bq, sc);
      }
#pragma unroll
      for (int r = 0; r < 8; ++r) {
        int key = kb + r + half * 8;
        float vv;
        if (key < kT) {
          int ck = coords[key];
          vv = sc[r] * scale + rel_bias(cq, ck, h, bias_rel, bias_demo);
        } else {
          vv = -INFINITY;
        }
        p[s * 8 + r] = vv;
        mloc = fmaxf(mloc, vv);
      }
    }
    mloc = fmaxf(mloc, __shfl_xor(mloc, 16, 32));
    float m_new = fmaxf(m_run, mloc);
    float alpha = __expf(m_run - m_new);
    float ls = 0.f;
#pragma unroll
    for (int i = 0; i < 16; ++i) { p[i] = __expf(p[i] - m_new); ls += p[i]; }
    l_run = l_run * alpha + ls + __shfl_xor(ls, 16, 32);
    m_run = m_new;
#pragma unroll
    for (int dt = 0; dt < 4; ++dt)
#pragma unroll
      for (int r = 0; r < 8; ++r) acc[dt][r] *= alpha;

    __syncthreads();
#pragma unroll
    for (int s = 0; s < 2; ++s)
#pragma unroll
      for (int r = 0; r < 8; ++r)
        lds_p[n * 32 + s * 16 + r + half * 8] = (_Float16)p[s * 8 + r];
    __syncthreads();

#pragma unroll
    for (int dt = 0; dt < 4; ++dt) {
      v16h a = load_a_frag(Vth + (size_t)(dt * 16) * kTP + k0, kTP, lane);  // Vt (d x keys)
      v16h bp = load_bT_frag(lds_p, 32, lane);                              // P^T (keys x q)
      acc[dt] = wmma_f16(a, bp, acc[dt]);
    }
    __syncthreads();
  }

  float inv = 1.f / l_run;
#pragma unroll
  for (int dt = 0; dt < 4; ++dt)
#pragma unroll
    for (int r = 0; r < 8; ++r) {
      int d = dt * 16 + r + half * 8;
      out[((size_t)b * kTP + q0 + n) * kD + h * kDH + d] = (_Float16)(acc[dt][r] * inv);
    }
}

// ---------------- head: logits = LN(x) @ Whead + bhead (V=12, VALU) ----------------
__global__ void head_kernel(const _Float16* __restrict__ xln,
                            const float* __restrict__ Wh,
                            const float* __restrict__ bh,
                            float* __restrict__ out) {
  int i = blockIdx.x * blockDim.x + threadIdx.x;
  int total = kB * kT * kV;
  if (i >= total) return;
  int v = i % kV;
  int bt = i / kV;
  int b = bt / kT, t = bt % kT;
  const _Float16* xr = xln + ((size_t)b * kTP + t) * kD;
  float acc = bh[v];
  for (int d = 0; d < kD; ++d) acc += (float)xr[d] * Wh[d * kV + v];
  out[i] = acc;
}

// ---------------- host orchestration ----------------
extern "C" void kernel_launch(void* const* d_in, const int* in_sizes, int n_in,
                              void* d_out, int out_size, void* d_ws, size_t ws_size,
                              hipStream_t stream) {
  (void)in_sizes; (void)n_in; (void)out_size; (void)ws_size;
  const int*   tokens   = (const int*)d_in[0];
  const float* tok_emb  = (const float*)d_in[1];
  const float* row_emb  = (const float*)d_in[2];
  const float* col_emb  = (const float*)d_in[3];
  const float* type_emb = (const float*)d_in[4];
  const float* bias_rel = (const float*)d_in[5];
  const float* bias_demo= (const float*)d_in[6];
  const float* ln1_s    = (const float*)d_in[7];
  const float* ln1_b    = (const float*)d_in[8];
  const float* Wqkv     = (const float*)d_in[9];
  const float* bqkv     = (const float*)d_in[10];
  const float* Wo       = (const float*)d_in[11];
  const float* bo       = (const float*)d_in[12];
  const float* ln2_s    = (const float*)d_in[13];
  const float* ln2_b    = (const float*)d_in[14];
  const float* W1       = (const float*)d_in[15];
  const float* b1       = (const float*)d_in[16];
  const float* W2       = (const float*)d_in[17];
  const float* b2       = (const float*)d_in[18];
  const float* lnf_s    = (const float*)d_in[19];
  const float* lnf_b    = (const float*)d_in[20];
  const float* Whead    = (const float*)d_in[21];
  const float* bhead    = (const float*)d_in[22];
  float* out = (float*)d_out;

  // workspace carve-up (all 256B aligned)
  char* base = (char*)d_ws;
  size_t off = 0;
  auto carve = [&](size_t bytes) {
    size_t o = off;
    off = (off + bytes + 255) & ~(size_t)255;
    return o;
  };
  int*       coords  = (int*)      (base + carve((size_t)kTP * 4));
  _Float16*  Wqkv16  = (_Float16*) (base + carve((size_t)kL * kD * 3 * kD * 2));  // (L, 3D, D)
  _Float16*  Wo16    = (_Float16*) (base + carve((size_t)kL * kD * kD * 2));      // (L, D, D)
  _Float16*  W116    = (_Float16*) (base + carve((size_t)kL * kD * kF * 2));      // (L, F, D)
  _Float16*  W216    = (_Float16*) (base + carve((size_t)kL * kF * kD * 2));      // (L, D, F)
  float*     x       = (float*)    (base + carve((size_t)kB * kTP * kD * 4));
  _Float16*  hbuf    = (_Float16*) (base + carve((size_t)kB * kTP * kD * 2));
  _Float16*  qkvb    = (_Float16*) (base + carve((size_t)kB * kTP * 3 * kD * 2));
  _Float16*  Vt      = (_Float16*) (base + carve((size_t)kB * kH * kDH * kTP * 2));
  _Float16*  attno   = (_Float16*) (base + carve((size_t)kB * kTP * kD * 2));
  _Float16*  mlph    = (_Float16*) (base + carve((size_t)kB * kTP * kF * 2));

  const int rows   = kB * kTP;       // 4032
  const int Mtiles = rows / 16;      // 252

  // 0) static coords + f16 transposed weight conversion
  coords_kernel<<<8, 256, 0, stream>>>(coords);
  {
    int n;
    n = kL * kD * 3 * kD;  cvtT_kernel<<<(n + 255) / 256, 256, 0, stream>>>(Wqkv, Wqkv16, kD, 3 * kD, n);
    n = kL * kD * kD;      cvtT_kernel<<<(n + 255) / 256, 256, 0, stream>>>(Wo,   Wo16,   kD, kD,     n);
    n = kL * kD * kF;      cvtT_kernel<<<(n + 255) / 256, 256, 0, stream>>>(W1,   W116,   kD, kF,     n);
    n = kL * kF * kD;      cvtT_kernel<<<(n + 255) / 256, 256, 0, stream>>>(W2,   W216,   kF, kD,     n);
  }

  // 1) embedding
  embed_kernel<<<rows, 32, 0, stream>>>(tokens, tok_emb, row_emb, col_emb, type_emb,
                                        coords, x);

  // 2) transformer layers
  for (int l = 0; l < kL; ++l) {
    // LN1 -> hbuf (f16)
    ln_kernel<<<rows, 32, 0, stream>>>(x, ln1_s + l * kD, ln1_b + l * kD, hbuf);
    // QKV = hbuf @ Wqkv[l] + bqkv[l]  -> f16 (rows x 1536)
    gemm16<0><<<dim3(Mtiles, 3 * kD / 64), 32, 0, stream>>>(
        hbuf, kD, Wqkv16 + (size_t)l * kD * 3 * kD, kD, bqkv + l * 3 * kD,
        nullptr, nullptr, qkvb, 3 * kD);
    // per-head V transpose
    vtrans_kernel<<<rows, 128, 0, stream>>>(qkvb, Vt);
    // attention with on-the-fly relative bias -> attno (f16)
    attn_kernel<<<dim3(kTP / 16, kH, kB), 32, 0, stream>>>(
        qkvb, Vt, coords, bias_rel, bias_demo, attno);
    // x = x + attno @ Wo[l] + bo[l]
    gemm16<0><<<dim3(Mtiles, kD / 64), 32, 0, stream>>>(
        attno, kD, Wo16 + (size_t)l * kD * kD, kD, bo + l * kD,
        x, x, nullptr, kD);
    // LN2 -> hbuf
    ln_kernel<<<rows, 32, 0, stream>>>(x, ln2_s + l * kD, ln2_b + l * kD, hbuf);
    // mlph = gelu(hbuf @ W1[l] + b1[l]) -> f16 (rows x 2048)
    gemm16<1><<<dim3(Mtiles, kF / 64), 32, 0, stream>>>(
        hbuf, kD, W116 + (size_t)l * kD * kF, kD, b1 + l * kF,
        nullptr, nullptr, mlph, kF);
    // x = x + mlph @ W2[l] + b2[l]
    gemm16<0><<<dim3(Mtiles, kD / 64), 32, 0, stream>>>(
        mlph, kF, W216 + (size_t)l * kF * kD, kF, b2 + l * kD,
        x, x, nullptr, kD);
  }

  // 3) final LN -> hbuf (f16), then head logits (B,T,V) fp32
  ln_kernel<<<rows, 32, 0, stream>>>(x, lnf_s, lnf_b, hbuf);
  {
    int total = kB * kT * kV;
    head_kernel<<<(total + 255) / 256, 256, 0, stream>>>(hbuf, Whead, bhead, out);
  }
}
